// MaskedSelfAttention_84593675862707
// MI455X (gfx1250) — matrix operand order repaired and checked by reading, not compiled, not run on hardware
//
#include <hip/hip_runtime.h>
#include <hip/hip_bf16.h>

typedef __attribute__((ext_vector_type(16))) _Float16 v16h;
typedef __attribute__((ext_vector_type(8)))  _Float16 v8h;
typedef __attribute__((ext_vector_type(2)))  __fp16   v2fp16;
typedef __attribute__((ext_vector_type(8)))  float    v8f;
typedef __attribute__((ext_vector_type(4)))  unsigned int u32x4;
typedef __attribute__((ext_vector_type(8)))  int      i32x8;
typedef __attribute__((ext_vector_type(4)))  int      i32x4;

#define EMBED 1024
#define SEQ   2048
#define BATCH 2
#define HEADS 16
#define HDIM  64
#define MTOT  (BATCH*SEQ)   // 4096

#if defined(__has_builtin)
#  if __has_builtin(__builtin_amdgcn_tensor_load_to_lds) && __has_builtin(__builtin_amdgcn_s_wait_tensorcnt)
#    define USE_TDM 1
#  endif
#endif
#ifndef USE_TDM
#  define USE_TDM 0
#endif

static __device__ __forceinline__ v8f wmma_f16(v16h a, v16h b, v8f c) {
  // D = A(16x32 f16) * B(32x16 f16) + C(16x16 f32)
  return __builtin_amdgcn_wmma_f32_16x16x32_f16(
      /*neg_a=*/false, a, /*neg_b=*/false, b,
      /*c_mod=*/(short)0, c, /*reuse_a=*/false, /*reuse_b=*/false);
}

static __device__ __forceinline__ v16h combine16(v8h lo, v8h hi) {
  union { v16h v; v8h h[2]; } u;
  u.h[0] = lo; u.h[1] = hi;
  return u.v;
}

// 16 contiguous f16 (32B) as two 16B chunks (LDS/global b128 friendly).
static __device__ __forceinline__ v16h load16(const _Float16* p) {
  const v8h* q = (const v8h*)p;
  return combine16(q[0], q[1]);
}

// v_cvt_pk_rtz_f16_f32: two f32 -> packed 2xf16 in one instruction
static __device__ __forceinline__ unsigned int pk_f32(float a, float b) {
  union { v2fp16 h; unsigned int u; } x;
  x.h = __builtin_amdgcn_cvt_pkrtz(a, b);
  return x.u;
}

// ---------------------------------------------------------------------------
// Tiled WMMA GEMM: C[4096,1024] = (A[4096,1024] @ W[1024,1024] + bias) * oscale
// AF16: 0 -> A is fp32 (pack-converted to f16 on stage), 1 -> A already f16
//   (AF16=1 uses the Tensor Data Mover to DMA the padded A tile into LDS)
// OUTMODE: 0 -> f16 scattered to [B,H,S,D]; 1 -> fp32 [M,N];
//          2 -> f16 scattered to [B,H,D,S] (transposed, for V)
// Block: 256 threads (8 waves), tile 128x256, k-step 32.
// Per wave: 64x64 output = 4x4 grid of 16x16 f32 accumulators (16 WMMA/k-step).
// ---------------------------------------------------------------------------
template<int AF16, int OUTMODE>
__global__ __launch_bounds__(256) void gemm_wmma_kernel(
    const void* __restrict__ Av, const float* __restrict__ W,
    const float* __restrict__ bias, void* __restrict__ outp, float oscale)
{
  __shared__ alignas(16) _Float16 sA [128][40];  // [m][k], 80B row stride
  __shared__ alignas(16) _Float16 sBT[256][40];  // [n][k] transposed, 80B stride

  const int tid  = threadIdx.x;
  const int m0   = blockIdx.x * 128;
  const int n0   = blockIdx.y * 256;
  const int wid  = tid >> 5;
  const int lane = tid & 31;
  const int lh   = lane & 15;        // 0..15
  const int hi   = lane >> 4;        // 0 or 1 (half-wave)
  const int wm   = (wid & 1) * 64;   // wave M offset in tile
  const int wn   = (wid >> 1) * 64;  // wave N offset in tile

  const v8f vzero = {};
  v8f acc[4][4];
  #pragma unroll
  for (int i = 0; i < 4; ++i)
    #pragma unroll
    for (int t = 0; t < 4; ++t) acc[i][t] = vzero;

  // staging assignments
  const int ar = tid >> 1, ab = (tid & 1) * 16;        // A: 128 rows x 32 k
  const int wk = (tid >> 4) * 2, nb = (tid & 15) * 16; // W: k-pairs x 16 n

  for (int ko = 0; ko < EMBED; ko += 32) {
    // ---- stage A tile ----
    if (AF16) {
#if USE_TDM
      // TDM: DMA the f16 tile [128 rows x 32 k] (row 64B) into LDS with
      // 16B padding after every row -> 80B LDS row stride (matches sA).
      if (wid == 0) {
        const unsigned long long ga =
            (unsigned long long)((const _Float16*)Av + (size_t)m0 * EMBED + ko);
        u32x4 g0;
        g0[0] = 1u;                                   // count=1 (valid), user D#
        g0[1] = (unsigned int)(size_t)&sA[0][0];      // LDS byte offset (low 32b)
        g0[2] = (unsigned int)(ga & 0xffffffffu);
        g0[3] = (unsigned int)((ga >> 32) & 0x1ffffffu) | (2u << 30); // type=2
        i32x8 g1;
        g1[0] = (int)((1u << 16)      // data_size = 2 bytes
                    | (1u << 20)      // pad_enable
                    | (3u << 22)      // pad_interval: every 16 DWORDs (64B)
                    | (3u << 25));    // pad_amount: 4 DWORDs (16B)
        g1[1] = (int)((unsigned)EMBED << 16);             // tensor_dim0 lo16
        g1[2] = (int)(((unsigned)EMBED >> 16) | ((unsigned)MTOT << 16)); // dim0 hi | dim1 lo
        g1[3] = (int)(((unsigned)MTOT >> 16) | (32u << 16));            // dim1 hi | tile_dim0=32
        g1[4] = 128;                                      // tile_dim1=128, tile_dim2=0
        g1[5] = EMBED;                                    // tensor_dim0_stride lo32
        g1[6] = 0;
        g1[7] = 0;
        const i32x4 gz  = {};
        const i32x8 gz8 = {};
        __builtin_amdgcn_tensor_load_to_lds(g0, g1, gz, gz, gz8, 0);
        __builtin_amdgcn_s_wait_tensorcnt(0);
      }
#else
      const v8h* src = (const v8h*)((const _Float16*)Av + (size_t)(m0 + ar) * EMBED + ko + ab);
      v8h a0 = src[0], a1 = src[1];
      *(v8h*)&sA[ar][ab]     = a0;
      *(v8h*)&sA[ar][ab + 8] = a1;
#endif
    } else {
      const float4* src = (const float4*)((const float*)Av + (size_t)(m0 + ar) * EMBED + ko + ab);
      unsigned int* dst = (unsigned int*)&sA[ar][ab];
      #pragma unroll
      for (int j = 0; j < 4; ++j) {
        float4 v = src[j];
        dst[2*j]     = pk_f32(v.x, v.y);
        dst[2*j + 1] = pk_f32(v.z, v.w);
      }
      if (ko + 32 < EMBED)
        __builtin_prefetch((const float*)Av + (size_t)(m0 + ar) * EMBED + ko + 32 + ab, 0, 0);
    }
    // ---- stage W tile transposed into [n][k]: pack k-pairs, 4B LDS stores ----
    {
      const float4* s0 = (const float4*)(W + (size_t)(ko + wk)     * EMBED + n0 + nb);
      const float4* s1 = (const float4*)(W + (size_t)(ko + wk + 1) * EMBED + n0 + nb);
      #pragma unroll
      for (int j = 0; j < 4; ++j) {
        float4 u0 = s0[j], u1 = s1[j];
        *(unsigned int*)&sBT[nb + 4*j + 0][wk] = pk_f32(u0.x, u1.x);
        *(unsigned int*)&sBT[nb + 4*j + 1][wk] = pk_f32(u0.y, u1.y);
        *(unsigned int*)&sBT[nb + 4*j + 2][wk] = pk_f32(u0.z, u1.z);
        *(unsigned int*)&sBT[nb + 4*j + 3][wk] = pk_f32(u0.w, u1.w);
      }
      if (ko + 32 < EMBED)
        __builtin_prefetch(W + (size_t)(ko + 32 + wk) * EMBED + n0 + nb, 0, 0);
    }
    __syncthreads();

    // ---- A fragments (16x32 A layout: lane<16 K{0..7,16..23}, lane>=16 K{8..15,24..31}) ----
    v16h af[4];
    #pragma unroll
    for (int i = 0; i < 4; ++i) {
      const _Float16* p = &sA[wm + 16*i + lh][hi * 8];
      af[i] = combine16(*(const v8h*)p, *(const v8h*)(p + 16));
    }
    // ---- B fragments (32x16: lane = column, lane<16 K{0..7,16..23}, lane>=16 K{8..15,24..31}) + WMMA ----
    #pragma unroll
    for (int t = 0; t < 4; ++t) {
      const _Float16* pb = &sBT[wn + 16*t + lh][hi * 8];
      const v16h bf = combine16(*(const v8h*)pb, *(const v8h*)(pb + 16));
      #pragma unroll
      for (int i = 0; i < 4; ++i) acc[i][t] = wmma_f16(af[i], bf, acc[i][t]);
    }
    __syncthreads();
  }

  // ---- epilogue: C layout lane=col, VGPR r = row (r / r+8 per half-wave) ----
  #pragma unroll
  for (int t = 0; t < 4; ++t) {
    const int col = n0 + wn + 16*t + lh;
    const float bv = bias[col];
    #pragma unroll
    for (int i = 0; i < 4; ++i) {
      #pragma unroll
      for (int r = 0; r < 8; ++r) {
        const int row = m0 + wm + 16*i + (hi ? r + 8 : r);
        const float v = (acc[i][t][r] + bv) * oscale;
        const int b = row >> 11, s = row & (SEQ - 1);
        const int h = col >> 6,  d = col & (HDIM - 1);
        if (OUTMODE == 0) {        // [B,H,S,D] f16
          ((_Float16*)outp)[((size_t)((b * HEADS + h) * SEQ + s)) * HDIM + d] = (_Float16)v;
        } else if (OUTMODE == 2) { // [B,H,D,S] f16 (transposed, for V)
          ((_Float16*)outp)[((size_t)((b * HEADS + h) * HDIM + d)) * SEQ + s] = (_Float16)v;
        } else {                   // fp32 [M,N]
          ((float*)outp)[(size_t)row * EMBED + col] = v;
        }
      }
    }
  }
}

// ---------------------------------------------------------------------------
// Flash attention (causal). Q,K in [B*H, S, D] f16 (Q pre-scaled by 1/sqrt(D));
// V in [B*H, D, S] f16 (transposed) so P@V B-fragments read contiguous rows
// straight from global/L2. Block: 128 threads = 4 independent waves, each
// owning 16 query rows; NO block barriers (only wave-private LDS P bounce).
// Row sums computed on the matrix pipe (P @ ones).
// ---------------------------------------------------------------------------
__global__ __launch_bounds__(128) void attn_kernel(
    const _Float16* __restrict__ Q, const _Float16* __restrict__ K,
    const _Float16* __restrict__ Vt, _Float16* __restrict__ O)
{
  __shared__ alignas(16) _Float16 sP[4][16][32];    // per-wave P tile (16q x 32k)

  const int bh  = blockIdx.y;             // b*HEADS + h
  const int q0  = blockIdx.x * 64;
  const int tid = threadIdx.x;
  const int wid = tid >> 5;
  const int lane = tid & 31;
  const int lh = lane & 15, hi = lane >> 4;
  const int qb = q0 + wid * 16;           // this wave's first query row

  const _Float16* Qb = Q  + (size_t)bh * SEQ * HDIM;
  const _Float16* Kb = K  + (size_t)bh * SEQ * HDIM;
  const _Float16* Vb = Vt + (size_t)bh * HDIM * SEQ;

  // Q fragments held for the whole key loop (A layout, D=64 -> 2 frags)
  v16h qf[2];
  #pragma unroll
  for (int f = 0; f < 2; ++f) {
    const _Float16* p = &Qb[(size_t)(qb + lh) * HDIM + 32*f + hi * 8];
    qf[f] = combine16(*(const v8h*)p, *(const v8h*)(p + 16));
  }
  // all-ones B fragment for matrix-pipe row sums
  v16h ones;
  #pragma unroll
  for (int i = 0; i < 16; ++i) ones[i] = (_Float16)1.0f;

  const v8f vzero = {};
  v8f acc[4] = {vzero, vzero, vzero, vzero};  // O tiles over d (4 x 16 cols)
  float mrow[8], lrow[8];
  #pragma unroll
  for (int r = 0; r < 8; ++r) { mrow[r] = -__builtin_inff(); lrow[r] = 0.0f; }

  const int nk = q0 / 32 + 2;       // key chunks for causal coverage

  for (int kt = 0; kt < nk; ++kt) {
    const int kb = kt * 32;
    // prefetch next K/V chunks while this one computes
    if (kb + 32 < SEQ) {
      __builtin_prefetch(Kb + (size_t)(kb + 32 + lane) * HDIM, 0, 0);
      __builtin_prefetch(Vb + (size_t)lane * SEQ + kb + 32, 0, 0);
      __builtin_prefetch(Vb + (size_t)(32 + lane) * SEQ + kb + 32, 0, 0);
    }

    // ---- scores = Q K^T (B-frags straight from global: K rows contiguous in d) ----
    v8f ct[2];
    #pragma unroll
    for (int kt2 = 0; kt2 < 2; ++kt2) {
      ct[kt2] = vzero;
      const int keyc = kb + kt2 * 16 + lh;   // this lane's key column
      #pragma unroll
      for (int f = 0; f < 2; ++f) {
        const _Float16* kp = &Kb[(size_t)keyc * HDIM + 32*f + hi * 8];
        const v16h bf = combine16(*(const v8h*)kp, *(const v8h*)(kp + 16));
        ct[kt2] = wmma_f16(qf[f], bf, ct[kt2]);
      }
    }

    // ---- causal mask only on diagonal chunks (wave-uniform branch) ----
    if (kb + 32 > qb) {
      #pragma unroll
      for (int r = 0; r < 8; ++r) {
        const int qr = qb + (hi ? r + 8 : r);
        if (kb + lh      > qr) ct[0][r] = -__builtin_inff();
        if (kb + 16 + lh > qr) ct[1][r] = -__builtin_inff();
      }
    }

    // ---- online softmax: row max via 16-lane shuffle reduce ----
    float fac[8];
    #pragma unroll
    for (int r = 0; r < 8; ++r) {
      float mx = fmaxf(ct[0][r], ct[1][r]);
      #pragma unroll
      for (int off = 1; off < 16; off <<= 1) mx = fmaxf(mx, __shfl_xor(mx, off, 32));
      const float nm = fmaxf(mrow[r], mx);
      fac[r] = __expf(mrow[r] - nm);
      const float p0 = __expf(ct[0][r] - nm);
      const float p1 = __expf(ct[1][r] - nm);
      mrow[r] = nm;
      const int rr = hi ? r + 8 : r;
      sP[wid][rr][lh]      = (_Float16)p0;      // C-layout -> row-major P in LDS
      sP[wid][rr][lh + 16] = (_Float16)p1;
      #pragma unroll
      for (int t = 0; t < 4; ++t) acc[t][r] *= fac[r];
    }
    // wave-private LDS; DS ops from one wave execute in order (no barrier).

    // ---- P A-frag from LDS; row sums on matrix pipe; P @ V^T rows ----
    const _Float16* pp = &sP[wid][lh][hi * 8];
    const v16h pa = combine16(*(const v8h*)pp, *(const v8h*)(pp + 16));
    const v8f rsv = wmma_f16(pa, ones, vzero);   // rsv[r] = rowsum of f16 P
    #pragma unroll
    for (int t = 0; t < 4; ++t) {
      const _Float16* vp = &Vb[(size_t)(16*t + lh) * SEQ + kb + hi * 8];
      const v16h vbf = combine16(*(const v8h*)vp, *(const v8h*)(vp + 16));
      acc[t] = wmma_f16(pa, vbf, acc[t]);
    }
    #pragma unroll
    for (int r = 0; r < 8; ++r) lrow[r] = lrow[r] * fac[r] + rsv[r];
  }

  // ---- normalize + write attention output to [B,S,E] f16 workspace ----
  const int b = bh >> 4, h = bh & 15;
  #pragma unroll
  for (int t = 0; t < 4; ++t) {
    const int col = h * HDIM + 16*t + lh;
    #pragma unroll
    for (int r = 0; r < 8; ++r) {
      const int qr = qb + (hi ? r + 8 : r);
      O[((size_t)(b * SEQ + qr)) * EMBED + col] = (_Float16)(acc[t][r] / lrow[r]);
    }
  }
}

// ---------------------------------------------------------------------------
extern "C" void kernel_launch(void* const* d_in, const int* in_sizes, int n_in,
                              void* d_out, int out_size, void* d_ws, size_t ws_size,
                              hipStream_t stream)
{
  (void)in_sizes; (void)n_in; (void)out_size; (void)ws_size;
  const float* x  = (const float*)d_in[0];
  const float* Wq = (const float*)d_in[1];
  const float* bq = (const float*)d_in[2];
  const float* Wk = (const float*)d_in[3];
  const float* bk = (const float*)d_in[4];
  const float* Wv = (const float*)d_in[5];
  const float* bv = (const float*)d_in[6];
  const float* Wo = (const float*)d_in[7];
  const float* bo = (const float*)d_in[8];
  float* out = (float*)d_out;

  // Workspace layout (f16): Q | K | V^T | attn_out  (8 MB each, 32 MB total)
  const size_t QKV = (size_t)BATCH * HEADS * SEQ * HDIM;
  _Float16* qws = (_Float16*)d_ws;
  _Float16* kws = qws + QKV;
  _Float16* vws = kws + QKV;
  _Float16* ows = vws + QKV;

  const dim3 gg(MTOT / 128, EMBED / 256);  // 32 x 4 blocks

  // Q gets the 1/sqrt(HDIM) scale folded into its projection; V stored transposed.
  gemm_wmma_kernel<0, 0><<<gg, dim3(256), 0, stream>>>((const void*)x, Wq, bq, (void*)qws, 0.125f);
  gemm_wmma_kernel<0, 0><<<gg, dim3(256), 0, stream>>>((const void*)x, Wk, bk, (void*)kws, 1.0f);
  gemm_wmma_kernel<0, 2><<<gg, dim3(256), 0, stream>>>((const void*)x, Wv, bv, (void*)vws, 1.0f);

  attn_kernel<<<dim3(SEQ / 64, BATCH * HEADS), dim3(128), 0, stream>>>(qws, kws, vws, ows);

  gemm_wmma_kernel<1, 1><<<gg, dim3(256), 0, stream>>>((const void*)ows, Wo, bo, (void*)out, 1.0f);
}
